// WaveNetModel_4080218931840
// MI455X (gfx1250) — compile-verified, matrix operand order checked by reading, not run on hardware
//
#include <hip/hip_runtime.h>
#include <hip/hip_bf16.h>

typedef __attribute__((ext_vector_type(16))) _Float16 v16h;
typedef __attribute__((ext_vector_type(8)))  _Float16 v8h;
typedef __attribute__((ext_vector_type(8)))  float    v8f;
typedef __attribute__((ext_vector_type(4)))  float    v4f;

#define CC   128
#define TT   32768
#define BB   4
#define LL   10
#define WSTRIDE 136                 // halfs per LDS weight row (128 + 8 pad = 272B)
#define TPB  512
#define WAVES 16
#define COLS_PER_WG 256
#define NWG ((BB * TT) / COLS_PER_WG)   // 512 workgroups

// ---------------- weight convert: f32 -> f16 ----------------
__global__ void cvt_f32_to_f16(const float* __restrict__ src,
                               _Float16* __restrict__ dst, int n) {
  int i = blockIdx.x * blockDim.x + threadIdx.x;
  if (i < n) dst[i] = (_Float16)src[i];
}

// -------- CDNA5 async global->LDS copy (ASYNCcnt path), GVS addressing --------
__device__ __forceinline__ void async_ld_b128(unsigned lds_byte_addr,
                                              unsigned voff_bytes,
                                              const void* sbase) {
  asm volatile("global_load_async_to_lds_b128 %0, %1, %2"
               :: "v"(lds_byte_addr), "v"(voff_bytes), "s"(sbase)
               : "memory");
}
__device__ __forceinline__ void wait_async0() {
  asm volatile("s_wait_asynccnt 0" ::: "memory");
}

// Load one 16x32 f16 A-fragment (W tile) from LDS (row-major, padded stride).
__device__ __forceinline__ v16h load_a(const _Float16* Wm, int m, int kc,
                                       int lane15, int hi) {
  const _Float16* p = Wm + (m * 16 + lane15) * WSTRIDE + kc * 32 + hi * 8;
  v8h lo = *(const v8h*)p;
  v8h hh = *(const v8h*)(p + 16);
  v16h a;
#pragma unroll
  for (int j = 0; j < 8; ++j) { a[j] = lo[j]; a[j + 8] = hh[j]; }
  return a;
}

__device__ __forceinline__ v8f bias8(const float* p) {
  v4f a = *(const v4f*)p;
  v4f b = *(const v4f*)(p + 4);
  v8f r;
#pragma unroll
  for (int j = 0; j < 4; ++j) { r[j] = a[j]; r[j + 4] = b[j]; }
  return r;
}

// D-fragments (8 x v8f) -> B-operand fragments (4 x v16h); pure in-lane repack.
__device__ __forceinline__ void d2b(const v8f* D, v16h* Bf) {
#pragma unroll
  for (int kc = 0; kc < 4; ++kc) {
    v16h t;
#pragma unroll
    for (int j = 0; j < 8; ++j) {
      t[j]     = (_Float16)D[2 * kc][j];
      t[j + 8] = (_Float16)D[2 * kc + 1][j];
    }
    Bf[kc] = t;
  }
}

__device__ __forceinline__ void d2b_relu(const v8f* D, v16h* Bf) {
#pragma unroll
  for (int kc = 0; kc < 4; ++kc) {
    v16h t;
#pragma unroll
    for (int j = 0; j < 8; ++j) {
      t[j]     = (_Float16)fmaxf(D[2 * kc][j], 0.0f);
      t[j + 8] = (_Float16)fmaxf(D[2 * kc + 1][j], 0.0f);
    }
    Bf[kc] = t;
  }
}

// acc[8] (128x16 f32) += W(128x128 f16 in LDS) * B(128x16 f16 frags)
__device__ __forceinline__ void gemm_acc(const _Float16* Wlds, const v16h* Bf,
                                         v8f* acc, int lane15, int hi) {
#pragma unroll
  for (int m = 0; m < 8; ++m) {
    v8f c = acc[m];
#pragma unroll
    for (int kc = 0; kc < 4; ++kc) {
      v16h a = load_a(Wlds, m, kc, lane15, hi);
      c = __builtin_amdgcn_wmma_f32_16x16x32_f16(false, a, false, Bf[kc],
                                                 (short)0, c, false, false);
    }
    acc[m] = c;
  }
}

// ---------------- fused 10-layer WaveNet stack + head ----------------
__global__ __launch_bounds__(TPB) void wavenet_main(
    const float* __restrict__ x,
    const void* __restrict__ wbase,           // d_ws base: all f16 weights
    const float* __restrict__ bd, const float* __restrict__ br,
    const float* __restrict__ bs,
    const float* __restrict__ bf1, const float* __restrict__ bf2,
    float* __restrict__ partial) {
  __shared__ _Float16 sW[2][3][128 * WSTRIDE];   // double-buffered: 2 x 104,448 B
  __shared__ float sBias[2][3][CC];
  __shared__ float sRed[WAVES][CC];

  const int tid    = threadIdx.x;
  const int lane   = tid & 31;
  const int wave   = tid >> 5;
  const int lane15 = lane & 15;
  const int hi     = lane >> 4;
  const int wg     = blockIdx.x;

  // ---- stage layer 0 weights asynchronously into buffer 0 ----
#pragma unroll
  for (int j = 0; j < 12; ++j) {
    int i = tid + j * TPB;                 // 3 * 2048 b128 chunks
    int mtx = i >> 11, idx = i & 2047;
    unsigned voff = (unsigned)((mtx * LL + 0) * (CC * CC) + idx * 8) * 2u;
    int row = idx >> 4, c8 = idx & 15;
    unsigned lds = (unsigned)(size_t)&sW[0][mtx][row * WSTRIDE + c8 * 8];
    async_ld_b128(lds, voff, wbase);
  }
  if (tid < 3 * CC) {
    int which = tid >> 7, c = tid & 127;
    const float* p = (which == 0) ? bd : (which == 1) ? br : bs;
    sBias[0][which][c] = p[c];             // layer 0 biases
  }

  // ---- load x tile into register D-fragments (overlaps async weight copy) ----
  const int col0 = wg * COLS_PER_WG + wave * 16;
  const int b    = col0 >> 15;                     // T = 32768
  const int t    = (col0 & (TT - 1)) + lane15;
  const float* xb = x + (size_t)b * CC * TT + t;

  v8f hD[8], skipD[8];
  const v8f vzero = {};
#pragma unroll
  for (int m = 0; m < 8; ++m) {
    v8f hv;
#pragma unroll
    for (int r = 0; r < 8; ++r)
      hv[r] = xb[(size_t)(m * 16 + hi * 8 + r) * TT];   // coalesced 64B/half-wave
    hD[m]    = hv;
    skipD[m] = vzero;
  }

  wait_async0();
  __syncthreads();

  for (int k = 0; k < LL; ++k) {
    const int cur = k & 1, nxt = cur ^ 1;

    // ---- prefetch next layer (or head) weights into the other buffer ----
    if (k + 1 < LL) {
#pragma unroll
      for (int j = 0; j < 12; ++j) {
        int i = tid + j * TPB;
        int mtx = i >> 11, idx = i & 2047;
        unsigned voff = (unsigned)((mtx * LL + (k + 1)) * (CC * CC) + idx * 8) * 2u;
        int row = idx >> 4, c8 = idx & 15;
        unsigned lds = (unsigned)(size_t)&sW[nxt][mtx][row * WSTRIDE + c8 * 8];
        async_ld_b128(lds, voff, wbase);
      }
      if (tid < 3 * CC) {
        int which = tid >> 7, c = tid & 127;
        const float* p = (which == 0) ? bd : (which == 1) ? br : bs;
        sBias[nxt][which][c] = p[(k + 1) * CC + c];
      }
    } else {  // head: Wf1, Wf2 at half-offset 3*LL*CC*CC
#pragma unroll
      for (int j = 0; j < 8; ++j) {
        int i = tid + j * TPB;                 // 2 * 2048 b128 chunks
        int mtx = i >> 11, idx = i & 2047;
        unsigned voff = (unsigned)((3 * LL + mtx) * (CC * CC) + idx * 8) * 2u;
        int row = idx >> 4, c8 = idx & 15;
        unsigned lds = (unsigned)(size_t)&sW[nxt][mtx][row * WSTRIDE + c8 * 8];
        async_ld_b128(lds, voff, wbase);
      }
      if (tid < 2 * CC) {
        int which = tid >> 7, c = tid & 127;
        sBias[nxt][which][c] = ((which == 0) ? bf1 : bf2)[c];
      }
    }

    // ---- compute layer k from current buffer ----
    v16h hB[4];
    d2b(hD, hB);

    v8f outD[8];                                    // out = Wd*h + bd
#pragma unroll
    for (int m = 0; m < 8; ++m) outD[m] = bias8(&sBias[cur][0][m * 16 + hi * 8]);
    gemm_acc(sW[cur][0], hB, outD, lane15, hi);

    v16h oB[4];
    d2b(outD, oB);

#pragma unroll                                      // skip += Ws*out + bs
    for (int m = 0; m < 8; ++m) skipD[m] += bias8(&sBias[cur][2][m * 16 + hi * 8]);
    gemm_acc(sW[cur][2], oB, skipD, lane15, hi);

#pragma unroll                                      // h = Wr*out + br + h
    for (int m = 0; m < 8; ++m) hD[m] += bias8(&sBias[cur][1][m * 16 + hi * 8]);
    gemm_acc(sW[cur][1], oB, hD, lane15, hi);

    wait_async0();      // my async copies for layer k+1 are done
    __syncthreads();    // everyone done copying nxt and done reading cur
  }

  // ---- head (weights staged in buffer 0 during layer 9) ----
  v16h fB[4];
  d2b_relu(skipD, fB);                    // relu(skip_sum)

  v8f gD[8];
#pragma unroll
  for (int m = 0; m < 8; ++m) gD[m] = bias8(&sBias[0][0][m * 16 + hi * 8]);
  gemm_acc(sW[0][0], fB, gD, lane15, hi); // Wf1 * f + bf1

  v16h gB[4];
  d2b_relu(gD, gB);                       // relu

  v8f yD[8];
#pragma unroll
  for (int m = 0; m < 8; ++m) yD[m] = bias8(&sBias[0][1][m * 16 + hi * 8]);
  gemm_acc(sW[0][1], gB, yD, lane15, hi); // Wf2 * g + bf2

  // deterministic partial mean: reduce this wave's 16 columns per row
#pragma unroll
  for (int m = 0; m < 8; ++m) {
#pragma unroll
    for (int r = 0; r < 8; ++r) {
      float v = yD[m][r];
      v += __shfl_xor(v, 1);
      v += __shfl_xor(v, 2);
      v += __shfl_xor(v, 4);
      v += __shfl_xor(v, 8);
      if (lane15 == 0) sRed[wave][m * 16 + hi * 8 + r] = v;
    }
  }
  __syncthreads();
  if (tid < CC) {
    float s = 0.f;
#pragma unroll
    for (int w = 0; w < WAVES; ++w) s += sRed[w][tid];
    partial[(size_t)wg * CC + tid] = s;   // fixed summation order -> deterministic
  }
}

// ---------------- final deterministic reduction over workgroups ----------------
__global__ void reduce_partials(const float* __restrict__ partial,
                                float* __restrict__ out) {
  int b = blockIdx.x, c = threadIdx.x;
  const int wpb = NWG / BB;               // 128 workgroups per batch element
  float s = 0.f;
  for (int w = 0; w < wpb; ++w)
    s += partial[(size_t)(b * wpb + w) * CC + c];
  out[b * CC + c] = s * (1.0f / TT);
}

extern "C" void kernel_launch(void* const* d_in, const int* in_sizes, int n_in,
                              void* d_out, int out_size, void* d_ws, size_t ws_size,
                              hipStream_t stream) {
  const float* x   = (const float*)d_in[0];
  const float* Wd  = (const float*)d_in[1];
  const float* bd  = (const float*)d_in[2];
  const float* Wr  = (const float*)d_in[3];
  const float* br  = (const float*)d_in[4];
  const float* Ws  = (const float*)d_in[5];
  const float* bs  = (const float*)d_in[6];
  const float* Wf1 = (const float*)d_in[7];
  const float* bf1 = (const float*)d_in[8];
  const float* Wf2 = (const float*)d_in[9];
  const float* bf2 = (const float*)d_in[10];

  // workspace layout: [f16 weights: exactly 1 MiB][f32 partials: 256 KiB]
  _Float16* wsH  = (_Float16*)d_ws;
  _Float16* Wdh  = wsH;
  _Float16* Wrh  = wsH + (size_t)1 * LL * CC * CC;
  _Float16* Wsh  = wsH + (size_t)2 * LL * CC * CC;
  _Float16* Wf1h = wsH + (size_t)3 * LL * CC * CC;
  _Float16* Wf2h = Wf1h + CC * CC;
  float* partial = (float*)((char*)d_ws + (1u << 20));

  const int nW = LL * CC * CC;
  cvt_f32_to_f16<<<(nW + 255) / 256, 256, 0, stream>>>(Wd, Wdh, nW);
  cvt_f32_to_f16<<<(nW + 255) / 256, 256, 0, stream>>>(Wr, Wrh, nW);
  cvt_f32_to_f16<<<(nW + 255) / 256, 256, 0, stream>>>(Ws, Wsh, nW);
  cvt_f32_to_f16<<<(CC * CC + 255) / 256, 256, 0, stream>>>(Wf1, Wf1h, CC * CC);
  cvt_f32_to_f16<<<(CC * CC + 255) / 256, 256, 0, stream>>>(Wf2, Wf2h, CC * CC);

  wavenet_main<<<NWG, TPB, 0, stream>>>(x, (const void*)d_ws, bd, br, bs,
                                        bf1, bf2, partial);
  reduce_partials<<<BB, CC, 0, stream>>>(partial, (float*)d_out);
}
